// SelfAttention_12953621365329
// MI455X (gfx1250) — compile-verified, matrix operand order checked by reading, not run on hardware
//
#include <hip/hip_runtime.h>
#include <math.h>

typedef __attribute__((ext_vector_type(16))) __bf16 v16bf;
typedef __attribute__((ext_vector_type(8)))  __bf16 v8bf;
typedef __attribute__((ext_vector_type(4)))  __bf16 v4bf;
typedef __attribute__((ext_vector_type(8)))  float  v8f;
typedef __attribute__((ext_vector_type(4)))  unsigned int v4u;
typedef __attribute__((ext_vector_type(8)))  int v8i;
typedef __attribute__((ext_vector_type(4)))  int v4i;

#define B_   4
#define N_   2048
#define H_   12
#define D_   64
#define DM_  768
#define BR   128   // query rows per workgroup (8 waves x 16)
#define BC   64    // key rows per chunk
#define NWAVE 8
// logits pre-scaled by 1/sqrt(64) * log2(e): softmax computed in base-2 domain
#define SCALE_LOG2E 0.1803368801111204f
#define EPS  1e-8f

// xor-butterfly within 16-lane groups via ds_swizzle (group-of-32: xor<<10 | and 0x1f)
#define SWZ_XOR(x, imm) __int_as_float(__builtin_amdgcn_ds_swizzle(__float_as_int(x), (imm)))

// ---- LDS carve (static shared block assumed at LDS offset 0) ----
#define RAWK_OFF   0        // 64x64 f32  (16 KB)
#define RAWV_OFF   16384    // 64x64 f32  (16 KB)
#define LDSK_OFF   32768    // 64x64 bf16 ( 8 KB)
#define LDSVT_OFF  40960    // 64x64 bf16 ( 8 KB, transposed [d][row])
#define LDSP_OFF   49152    // 8 waves x 16 x 64 bf16 (16 KB)
#define SMEM_BYTES 65536

// Issue a TDM 2D tile load: tile 64(d) x 64(rows) of f32, row stride DM_,
// packed contiguously into LDS at lds_off. 6-arg toolchain builtin.
__device__ __forceinline__ void tdm_load_2d(unsigned lds_off, const float* gaddr) {
    const unsigned long long ga = (unsigned long long)(uintptr_t)gaddr;
    v4u g0;
    g0[0] = 1u;                                            // count=1 (valid user D#)
    g0[1] = lds_off;                                       // lds_addr (bytes)
    g0[2] = (unsigned)(ga & 0xFFFFFFFFu);                  // global_addr[31:0]
    g0[3] = (unsigned)((ga >> 32) & 0x01FFFFFFu)           // global_addr[56:32]
          | (2u << 30);                                    // type=2 ("image")
    v8i g1;
    g1[0] = (int)(2u << 16);                               // data_size=4B; wg_mask=0
    g1[1] = (int)(((unsigned)D_ & 0xFFFFu) << 16);         // tensor_dim0[15:0] @63:48
    g1[2] = (int)(((unsigned)N_ & 0xFFFFu) << 16);         // dim0[31:16]=0 | tensor_dim1[15:0]
    g1[3] = (int)(((unsigned)D_ & 0xFFFFu) << 16);         // dim1[31:16]=0 | tile_dim0=64
    g1[4] = (int)((unsigned)BC);                           // tile_dim1=64 | tile_dim2=0
    g1[5] = (int)DM_;                                      // tensor_dim0_stride[31:0]
    g1[6] = 0;                                             // stride0[47:32] | stride1[15:0]
    g1[7] = 0;                                             // stride1[47:16]
    const v4i z4 = {0, 0, 0, 0};
    const v8i z8 = {0, 0, 0, 0, 0, 0, 0, 0};
    __builtin_amdgcn_tensor_load_to_lds(g0, g1, z4, z4, z8, 0);
}

__global__ __launch_bounds__(256)
void SelfAttention_fa_bf16_wmma(const float* __restrict__ K,
                                const float* __restrict__ Q,
                                const float* __restrict__ V,
                                float* __restrict__ O)
{
    __shared__ __align__(16) unsigned char smem[SMEM_BYTES];
    float*  rawK   = (float*)(smem + RAWK_OFF);    // TDM destination (f32)
    float*  rawV   = (float*)(smem + RAWV_OFF);
    __bf16* lds_k  = (__bf16*)(smem + LDSK_OFF);   // [row][d]
    __bf16* lds_vT = (__bf16*)(smem + LDSVT_OFF);  // [d][row]
    __bf16* lds_p  = (__bf16*)(smem + LDSP_OFF);   // [wave][M][col]

    const int tid  = threadIdx.x;
    const int w    = tid >> 5;       // wave id 0..7
    const int lane = tid & 31;
    const int lo   = lane & 15;      // N / M index within 16
    const int hi   = lane >> 4;      // half-wave selector

    const int b = blockIdx.z;
    const int h = blockIdx.y;
    const int qbase = blockIdx.x * BR + w * 16;

    const float* Qp = Q + (size_t)b * N_ * DM_ + (size_t)h * D_;
    const float* Kp = K + (size_t)b * N_ * DM_ + (size_t)h * D_;
    const float* Vp = V + (size_t)b * N_ * DM_ + (size_t)h * D_;
    float*       Op = O + (size_t)b * N_ * DM_ + (size_t)h * D_;

    // ---- Q fragments (A layout: lane M = lo; elem e -> K = (e>=8?16:0)+hi*8+(e&7)) ----
    v16bf aq[2];
    {
        const float* qrow = Qp + (size_t)(qbase + lo) * DM_;
        #pragma unroll
        for (int dh = 0; dh < 2; ++dh)
            #pragma unroll
            for (int g = 0; g < 2; ++g) {
                const int dstart = dh * 32 + g * 16 + hi * 8;
                #pragma unroll
                for (int j = 0; j < 8; ++j)
                    aq[dh][g * 8 + j] = (__bf16)(qrow[dstart + j] * SCALE_LOG2E);
            }
    }

    // all-ones B fragment: P x ones computes row-sums of P on the tensor core
    v16bf bones;
    #pragma unroll
    for (int j = 0; j < 16; ++j) bones[j] = (__bf16)1.0f;

    v8f o[4] = {v8f{}, v8f{}, v8f{}, v8f{}};
    v8f osum = {};
    float mrow[8];
    #pragma unroll
    for (int r = 0; r < 8; ++r) mrow[r] = -INFINITY;

    // ---- kick off TDM for chunk 0 (wave 0 only; EXEC-independent wave-level op) ----
    if (w == 0) {
        tdm_load_2d(RAWK_OFF, Kp);
        tdm_load_2d(RAWV_OFF, Vp);
    }

    for (int j0 = 0; j0 < N_; j0 += BC) {
        if (w == 0) __builtin_amdgcn_s_wait_tensorcnt(0);  // chunk j0 resident in LDS
        __syncthreads();                                   // (A) raw ready; prev compute done

        // ---- LDS convert pass: f32 -> bf16, V transposed ----
        #pragma unroll
        for (int e = tid; e < (BC * D_) / 4; e += 256) {
            const int row = e >> 4;            // keyrow within chunk
            const int col = (e & 15) << 2;     // d, float4 granularity
            const float4 kq = *(const float4*)&rawK[row * D_ + col];
            const float4 vq = *(const float4*)&rawV[row * D_ + col];
            v4bf kb = { (__bf16)kq.x, (__bf16)kq.y, (__bf16)kq.z, (__bf16)kq.w };
            *(v4bf*)&lds_k[row * D_ + col] = kb;
            lds_vT[(col + 0) * BC + row] = (__bf16)vq.x;
            lds_vT[(col + 1) * BC + row] = (__bf16)vq.y;
            lds_vT[(col + 2) * BC + row] = (__bf16)vq.z;
            lds_vT[(col + 3) * BC + row] = (__bf16)vq.w;
        }
        __syncthreads();                                   // (B) bf16 operands ready

        // ---- overlap: DMA next chunk while this chunk computes ----
        if (w == 0 && j0 + BC < N_) {
            tdm_load_2d(RAWK_OFF, Kp + (size_t)(j0 + BC) * DM_);
            tdm_load_2d(RAWV_OFF, Vp + (size_t)(j0 + BC) * DM_);
        }

        // ---- S = (Q*scale*log2e) K^T : four 16x16 f32 tiles over 64 key rows ----
        v8f c[4] = {v8f{}, v8f{}, v8f{}, v8f{}};
        #pragma unroll
        for (int nc = 0; nc < 4; ++nc) {
            const int row = nc * 16 + lo;
            #pragma unroll
            for (int dh = 0; dh < 2; ++dh) {
                v8bf g0 = *(const v8bf*)&lds_k[row * D_ + dh * 32 + hi * 8];
                v8bf g1 = *(const v8bf*)&lds_k[row * D_ + dh * 32 + 16 + hi * 8];
                v16bf bk = __builtin_shufflevector(g0, g1, 0,1,2,3,4,5,6,7,8,9,10,11,12,13,14,15);
                c[nc] = __builtin_amdgcn_wmma_f32_16x16x32_bf16(
                            false, aq[dh], false, bk, (short)0, c[nc], false, false);
            }
        }

        // ---- online softmax (base-2): row max over 64 cols, 16-lane xor butterfly ----
        float alpha[8];
        #pragma unroll
        for (int r = 0; r < 8; ++r) {
            float mx = fmaxf(fmaxf(c[0][r], c[1][r]), fmaxf(c[2][r], c[3][r]));
            mx = fmaxf(mx, SWZ_XOR(mx, 0x041f));   // xor 1
            mx = fmaxf(mx, SWZ_XOR(mx, 0x081f));   // xor 2
            mx = fmaxf(mx, SWZ_XOR(mx, 0x101f));   // xor 4
            mx = fmaxf(mx, SWZ_XOR(mx, 0x201f));   // xor 8
            const float mn = fmaxf(mrow[r], mx);
            alpha[r] = __builtin_amdgcn_exp2f(mrow[r] - mn);
            mrow[r]  = mn;
            const int M = r + hi * 8;
            #pragma unroll
            for (int nc = 0; nc < 4; ++nc)
                lds_p[(w * 16 + M) * BC + nc * 16 + lo] =
                    (__bf16)__builtin_amdgcn_exp2f(c[nc][r] - mn);
        }

        // ---- rescale O and row-sum accumulators by alpha ----
        #pragma unroll
        for (int r = 0; r < 8; ++r) {
            o[0][r] *= alpha[r]; o[1][r] *= alpha[r];
            o[2][r] *= alpha[r]; o[3][r] *= alpha[r];
            osum[r] *= alpha[r];
        }
        asm volatile("s_wait_dscnt 0" ::: "memory");   // wave-local LDS RAW fence

        // ---- reload P in A-matrix layout: two 16x32 bf16 frags ----
        v16bf ap0, ap1;
        {
            const __bf16* pr = &lds_p[(w * 16 + lo) * BC];
            v8bf g0 = *(const v8bf*)&pr[hi * 8];
            v8bf g1 = *(const v8bf*)&pr[16 + hi * 8];
            ap0 = __builtin_shufflevector(g0, g1, 0,1,2,3,4,5,6,7,8,9,10,11,12,13,14,15);
            v8bf g2 = *(const v8bf*)&pr[32 + hi * 8];
            v8bf g3 = *(const v8bf*)&pr[48 + hi * 8];
            ap1 = __builtin_shufflevector(g2, g3, 0,1,2,3,4,5,6,7,8,9,10,11,12,13,14,15);
        }

        // ---- row-sum of P via tensor core: osum += P x ones ----
        osum = __builtin_amdgcn_wmma_f32_16x16x32_bf16(false, ap0, false, bones, (short)0, osum, false, false);
        osum = __builtin_amdgcn_wmma_f32_16x16x32_bf16(false, ap1, false, bones, (short)0, osum, false, false);

        // ---- O += P V : four 16x16 d-tiles, K=64 in two steps ----
        #pragma unroll
        for (int t = 0; t < 4; ++t) {
            const int col = t * 16 + lo;          // d column (B-matrix N index)
            v8bf g0 = *(const v8bf*)&lds_vT[col * BC + hi * 8];
            v8bf g1 = *(const v8bf*)&lds_vT[col * BC + 16 + hi * 8];
            v16bf bv0 = __builtin_shufflevector(g0, g1, 0,1,2,3,4,5,6,7,8,9,10,11,12,13,14,15);
            v8bf g2 = *(const v8bf*)&lds_vT[col * BC + 32 + hi * 8];
            v8bf g3 = *(const v8bf*)&lds_vT[col * BC + 48 + hi * 8];
            v16bf bv1 = __builtin_shufflevector(g2, g3, 0,1,2,3,4,5,6,7,8,9,10,11,12,13,14,15);
            o[t] = __builtin_amdgcn_wmma_f32_16x16x32_bf16(false, ap0, false, bv0, (short)0, o[t], false, false);
            o[t] = __builtin_amdgcn_wmma_f32_16x16x32_bf16(false, ap1, false, bv1, (short)0, o[t], false, false);
        }
    }

    // ---- epilogue: divide by (l + eps), write fp32 output ----
    #pragma unroll
    for (int r = 0; r < 8; ++r) {
        const float inv = 1.0f / (osum[r] + EPS);
        float* orow = Op + (size_t)(qbase + r + hi * 8) * DM_;
        orow[ 0 + lo] = o[0][r] * inv;
        orow[16 + lo] = o[1][r] * inv;
        orow[32 + lo] = o[2][r] * inv;
        orow[48 + lo] = o[3][r] * inv;
    }
}

extern "C" void kernel_launch(void* const* d_in, const int* in_sizes, int n_in,
                              void* d_out, int out_size, void* d_ws, size_t ws_size,
                              hipStream_t stream) {
    // setup_inputs() dict order: k, q, v
    const float* k = (const float*)d_in[0];
    const float* q = (const float*)d_in[1];
    const float* v = (const float*)d_in[2];
    float* out = (float*)d_out;
    (void)in_sizes; (void)n_in; (void)out_size; (void)d_ws; (void)ws_size;

    dim3 grid(N_ / BR, H_, B_);   // (16, 12, 4)
    dim3 block(256);
    SelfAttention_fa_bf16_wmma<<<grid, block, 0, stream>>>(k, q, v, out);
}
